// MultiHeadLatentAttention_89197880803830
// MI455X (gfx1250) — compile-verified, matrix operand order checked by reading, not run on hardware
//
#include <hip/hip_runtime.h>
#include <hip/hip_bf16.h>

#define S_LEN   2048
#define D_MODEL 1024
#define NHEAD   16
#define HDIM    64
#define LAT     512
#define BATCH   4
#define NTOK    (BATCH * S_LEN)

typedef __attribute__((ext_vector_type(16))) __bf16 v16bf;
typedef __attribute__((ext_vector_type(8)))  float  v8f;
typedef unsigned short bf16raw;

union FragU { uint4 u[2]; v16bf v; };

__device__ __forceinline__ bf16raw f2bf(float f) {
    unsigned u = __builtin_bit_cast(unsigned, f);
    u += 0x7FFFu + ((u >> 16) & 1u);            // round-to-nearest-even
    return (bf16raw)(u >> 16);
}

// LDS byte offset of a __shared__ object: the LDS aperture occupies flat
// addr[63:32], so addr[31:0] is the wave-relative LDS address (ISA 10.2).
__device__ __forceinline__ unsigned lds_off(const void* p) {
    return (unsigned)(unsigned long long)p;
}

// Async DMA: global -> LDS, 16B per lane, tracked by ASYNCcnt (ISA 10.x/15.18).
__device__ __forceinline__ void async_copy_b128(unsigned lds, const void* gaddr) {
    asm volatile("global_load_async_to_lds_b128 %0, %1, off"
                 :: "v"(lds), "v"(gaddr) : "memory");
}

// Load one 16x32 bf16 WMMA operand fragment (A-pattern; B uses the same
// pattern against a transposed operand). Per ISA 7.12.2:
//   lanes 0-15 : row = base+lane,    K = {k0..k0+7, k0+16..k0+23}
//   lanes 16-31: row = base+lane-16, K = {k0+8..k0+15, k0+24..k0+31}
__device__ __forceinline__ v16bf load_frag(const bf16raw* base, int ld,
                                           int row16, int k0, int lane) {
    int r    = row16 + (lane & 15);
    int koff = k0 + ((lane >> 4) << 3);
    const bf16raw* p = base + (size_t)r * ld + koff;
    FragU f;
    f.u[0] = *(const uint4*)(p);
    f.u[1] = *(const uint4*)(p + 16);
    return f.v;
}

__device__ __forceinline__ v8f wmma_bf16(v16bf a, v16bf b, v8f c) {
    return __builtin_amdgcn_wmma_f32_16x16x32_bf16(false, a, false, b,
                                                   (short)0, c, false, false);
}

// ---------------------------------------------------------------- RMSNorm ---
__global__ __launch_bounds__(256)
void rmsnorm_kernel(const float* __restrict__ x, const float* __restrict__ w,
                    bf16raw* __restrict__ xn) {
    int t = blockIdx.x;
    const float* xr = x + (size_t)t * D_MODEL;
    float4 v = ((const float4*)xr)[threadIdx.x];
    float ss = v.x * v.x + v.y * v.y + v.z * v.z + v.w * v.w;
    #pragma unroll
    for (int m = 16; m >= 1; m >>= 1) ss += __shfl_xor(ss, m, 32);
    __shared__ float red[8];
    if ((threadIdx.x & 31) == 0) red[threadIdx.x >> 5] = ss;
    __syncthreads();
    float tot = 0.f;
    #pragma unroll
    for (int i = 0; i < 8; ++i) tot += red[i];
    float r = rsqrtf(tot * (1.0f / D_MODEL) + 1e-6f);
    float4 wv = ((const float4*)w)[threadIdx.x];
    bf16raw* o = xn + (size_t)t * D_MODEL + threadIdx.x * 4;
    o[0] = f2bf(v.x * r * wv.x);
    o[1] = f2bf(v.y * r * wv.y);
    o[2] = f2bf(v.z * r * wv.z);
    o[3] = f2bf(v.w * r * wv.w);
}

// ----------------------------------------- weight cast + transpose to bf16 --
__global__ __launch_bounds__(256)
void cast_transpose_kernel(const float* __restrict__ W, bf16raw* __restrict__ Wt,
                           int K, int N) {
    int idx = blockIdx.x * blockDim.x + threadIdx.x;
    if (idx >= K * N) return;
    int n = idx % N, k = idx / N;                 // coalesced read of W[k][n]
    Wt[(size_t)n * K + k] = f2bf(W[idx]);
}

// ------------------------------------------------------------- RoPE tables --
__global__ __launch_bounds__(256)
void rope_table_kernel(float* __restrict__ cosT, float* __restrict__ sinT) {
    int idx = blockIdx.x * blockDim.x + threadIdx.x;
    if (idx >= S_LEN * 32) return;
    int s = idx >> 5, j = idx & 31;
    float inv = __expf(-(2.0f * (float)j / (float)HDIM) * logf(10000.0f));
    float a = (float)s * inv;
    cosT[idx] = cosf(a);
    sinT[idx] = sinf(a);
}

// ------------------------------------------------------------ tiled GEMM ----
// C[M,N] = A[M,K](bf16, row-major) x Wt[N,K](bf16, pre-transposed)
// Block: 128x128 tile, 256 threads = 8 waves, wave = 32(M) x 64(N).
// K-slices stream through double-buffered LDS via async-to-LDS DMA so the
// tile(t+1) copy overlaps the tile(t) WMMA work.
// EPI: 0 = bf16 row-major, 1 = RoPE + bf16 row-major (N-slices head-aligned),
//      2 = f32 + residual,  3 = bf16 V-transposed (B,H,HD,S)
template<int EPI>
__global__ __launch_bounds__(256)
void gemm_kernel(const bf16raw* __restrict__ A, const bf16raw* __restrict__ Bt,
                 int M, int N, int K,
                 bf16raw* __restrict__ outb, float* __restrict__ outf,
                 const float* __restrict__ resid,
                 const float* __restrict__ cosT, const float* __restrict__ sinT) {
    __shared__ bf16raw Al[2][128 * 40];   // 32-K slice, +8 halfs pad, x2 buffers
    __shared__ bf16raw Bl[2][128 * 40];
    const int tid  = threadIdx.x;
    const int lane = tid & 31;
    const int w    = tid >> 5;
    const int wm   = w & 3;            // 32-row slice
    const int wn   = w >> 2;           // 64-col slice
    const int m0   = blockIdx.y * 128;
    const int n0   = blockIdx.x * 128;

    v8f acc[2][4] = {};

    // 4 async-to-LDS instructions per wave per tile (ASYNCcnt += 4)
    auto issue_tile = [&](int k0, int buf) {
        #pragma unroll
        for (int i = 0; i < 2; ++i) {
            int idx = tid + i * 256;
            int row = idx >> 2, seg = idx & 3;
            async_copy_b128(lds_off(&Al[buf][row * 40 + seg * 8]),
                            A + (size_t)(m0 + row) * K + k0 + seg * 8);
            async_copy_b128(lds_off(&Bl[buf][row * 40 + seg * 8]),
                            Bt + (size_t)(n0 + row) * K + k0 + seg * 8);
        }
    };

    const int T = K / 32;
    issue_tile(0, 0);
    for (int t = 0; t < T; ++t) {
        if (t + 1 < T) {
            issue_tile((t + 1) * 32, (t + 1) & 1);
            asm volatile("s_wait_asynccnt 4" ::: "memory");   // tile t landed
        } else {
            asm volatile("s_wait_asynccnt 0" ::: "memory");
        }
        __syncthreads();
        const bf16raw* Ab = Al[t & 1];
        const bf16raw* Bb = Bl[t & 1];
        v16bf af[2], bfr[4];
        #pragma unroll
        for (int mt = 0; mt < 2; ++mt)
            af[mt] = load_frag(Ab, 40, wm * 32 + mt * 16, 0, lane);
        #pragma unroll
        for (int nt = 0; nt < 4; ++nt)
            bfr[nt] = load_frag(Bb, 40, wn * 64 + nt * 16, 0, lane);
        #pragma unroll
        for (int mt = 0; mt < 2; ++mt)
            #pragma unroll
            for (int nt = 0; nt < 4; ++nt)
                acc[mt][nt] = wmma_bf16(af[mt], bfr[nt], acc[mt][nt]);
        __syncthreads();   // buffer (t&1) free for reuse at tile t+2
    }

    const int rsel = (lane >> 4) << 3;   // C-layout: lanes 16-31 hold rows +8
    if constexpr (EPI == 0 || EPI == 2 || EPI == 3) {
        #pragma unroll
        for (int mt = 0; mt < 2; ++mt)
            #pragma unroll
            for (int nt = 0; nt < 4; ++nt) {
                int col  = n0 + wn * 64 + nt * 16 + (lane & 15);
                int rowb = m0 + wm * 32 + mt * 16 + rsel;
                #pragma unroll
                for (int r = 0; r < 8; ++r) {
                    int row = rowb + r;
                    float v = acc[mt][nt][r];
                    if constexpr (EPI == 0) {
                        outb[(size_t)row * N + col] = f2bf(v);
                    } else if constexpr (EPI == 2) {
                        size_t o = (size_t)row * N + col;
                        outf[o] = v + resid[o];
                    } else {          // EPI == 3: (B,H,HD,S) transposed V
                        int b = row >> 11, s = row & (S_LEN - 1);
                        int h = col >> 6,  hd = col & 63;
                        outb[(((size_t)(b * NHEAD + h)) * HDIM + hd) * S_LEN + s] = f2bf(v);
                    }
                }
            }
    } else {  // EPI == 1: fused RoPE. Wave's 64-col slice is one whole head.
        #pragma unroll
        for (int mt = 0; mt < 2; ++mt)
            #pragma unroll
            for (int ntp = 0; ntp < 2; ++ntp) {
                int d    = ntp * 16 + (lane & 15);      // rotation index 0..31
                int colA = n0 + wn * 64 + d;
                int rowb = m0 + wm * 32 + mt * 16 + rsel;
                #pragma unroll
                for (int r = 0; r < 8; ++r) {
                    int row = rowb + r;
                    int s   = row & (S_LEN - 1);
                    float c  = cosT[s * 32 + d];
                    float sn = sinT[s * 32 + d];
                    float x1 = acc[mt][ntp][r];
                    float x2 = acc[mt][ntp + 2][r];
                    outb[(size_t)row * N + colA]      = f2bf(x1 * c - x2 * sn);
                    outb[(size_t)row * N + colA + 32] = f2bf(x1 * sn + x2 * c);
                }
            }
    }
}

// ----------------------------------------------------- flash attention ------
// Per block: one (b,h), 128 queries (8 waves x 16 rows). 64-key tiles stream
// through double-buffered LDS via async-to-LDS DMA.
__global__ __launch_bounds__(256)
void attn_kernel(const bf16raw* __restrict__ q, const bf16raw* __restrict__ k,
                 const bf16raw* __restrict__ vt, bf16raw* __restrict__ aout) {
    __shared__ bf16raw kb[2][64 * 72];     // 64 keys x 64 dims (+pad), x2
    __shared__ bf16raw vb[2][64 * 72];     // 64 dims x 64 keys (V^T tile), x2
    __shared__ bf16raw pb[8 * 16 * 72];    // per-wave P transpose scratch
    const int tid  = threadIdx.x;
    const int lane = tid & 31;
    const int w    = tid >> 5;
    const int bh   = blockIdx.y;
    const int b    = bh >> 4, h = bh & 15;
    const int q0   = blockIdx.x * 128 + w * 16;
    const int rsel = (lane >> 4) << 3;

    // Q fragments straight from global (A-pattern rows are contiguous b128s)
    const bf16raw* qbase = q + ((size_t)(b * S_LEN + q0)) * D_MODEL + h * HDIM;
    v16bf aq[2];
    #pragma unroll
    for (int kf = 0; kf < 2; ++kf) aq[kf] = load_frag(qbase, D_MODEL, 0, kf * 32, lane);

    v8f oacc[4] = {};
    float mrow[8], lrow[8];
    #pragma unroll
    for (int r = 0; r < 8; ++r) { mrow[r] = -3.0e38f; lrow[r] = 0.f; }
    bf16raw* pwave = pb + w * (16 * 72);

    // 4 async-to-LDS instructions per wave per key tile
    auto issue_kv = [&](int s0, int buf) {
        #pragma unroll
        for (int i = 0; i < 2; ++i) {
            int idx = tid + i * 256;
            int row = idx >> 3, seg = idx & 7;
            async_copy_b128(lds_off(&kb[buf][row * 72 + seg * 8]),
                            k + ((size_t)(b * S_LEN + s0 + row)) * D_MODEL
                              + h * HDIM + seg * 8);
            async_copy_b128(lds_off(&vb[buf][row * 72 + seg * 8]),
                            vt + ((size_t)(b * NHEAD + h) * HDIM + row) * S_LEN
                               + s0 + seg * 8);
        }
    };

    const int T = S_LEN / 64;
    issue_kv(0, 0);
    for (int it = 0; it < T; ++it) {
        if (it + 1 < T) {
            issue_kv((it + 1) * 64, (it + 1) & 1);
            asm volatile("s_wait_asynccnt 4" ::: "memory");
        } else {
            asm volatile("s_wait_asynccnt 0" ::: "memory");
        }
        __syncthreads();
        const bf16raw* kbuf = kb[it & 1];
        const bf16raw* vbuf = vb[it & 1];

        // scores: 16 q-rows x 64 keys per wave
        v8f sacc[4] = {};
        #pragma unroll
        for (int kf = 0; kf < 2; ++kf) {
            v16bf bk[4];
            #pragma unroll
            for (int nt = 0; nt < 4; ++nt)
                bk[nt] = load_frag(kbuf, 72, nt * 16, kf * 32, lane);
            #pragma unroll
            for (int nt = 0; nt < 4; ++nt)
                sacc[nt] = wmma_bf16(aq[kf], bk[nt], sacc[nt]);
        }

        // online softmax (row stats per VGPR slot; cross-lane within 16-group)
        #pragma unroll
        for (int r = 0; r < 8; ++r) {
            float mx = -3.0e38f;
            #pragma unroll
            for (int nt = 0; nt < 4; ++nt) mx = fmaxf(mx, sacc[nt][r]);
            #pragma unroll
            for (int msk = 8; msk >= 1; msk >>= 1)
                mx = fmaxf(mx, __shfl_xor(mx, msk, 16));
            mx *= 0.125f;                         // 1/sqrt(HD), order-preserving
            float mnew  = fmaxf(mrow[r], mx);
            float scale = __expf(mrow[r] - mnew);
            float lsum  = 0.f;
            #pragma unroll
            for (int nt = 0; nt < 4; ++nt) {
                float p = __expf(sacc[nt][r] * 0.125f - mnew);
                sacc[nt][r] = p;
                lsum += p;
            }
            #pragma unroll
            for (int msk = 8; msk >= 1; msk >>= 1)
                lsum += __shfl_xor(lsum, msk, 16);
            lrow[r] = lrow[r] * scale + lsum;
            mrow[r] = mnew;
            #pragma unroll
            for (int dt = 0; dt < 4; ++dt) oacc[dt][r] *= scale;
        }

        // C-layout P -> A-layout via per-wave LDS (DS ops in-order per wave)
        #pragma unroll
        for (int nt = 0; nt < 4; ++nt)
            #pragma unroll
            for (int r = 0; r < 8; ++r)
                pwave[(rsel + r) * 72 + nt * 16 + (lane & 15)] = f2bf(sacc[nt][r]);
        asm volatile("s_wait_dscnt 0" ::: "memory");

        v16bf pf[2];
        #pragma unroll
        for (int kf = 0; kf < 2; ++kf) pf[kf] = load_frag(pwave, 72, 0, kf * 32, lane);
        #pragma unroll
        for (int kf = 0; kf < 2; ++kf) {
            v16bf bv[4];
            #pragma unroll
            for (int dt = 0; dt < 4; ++dt)
                bv[dt] = load_frag(vbuf, 72, dt * 16, kf * 32, lane);
            #pragma unroll
            for (int dt = 0; dt < 4; ++dt)
                oacc[dt] = wmma_bf16(pf[kf], bv[dt], oacc[dt]);
        }
        __syncthreads();   // buffer (it&1) free for reuse at tile it+2
    }

    #pragma unroll
    for (int dt = 0; dt < 4; ++dt) {
        int f = h * HDIM + dt * 16 + (lane & 15);
        #pragma unroll
        for (int r = 0; r < 8; ++r) {
            int t = b * S_LEN + q0 + rsel + r;
            aout[(size_t)t * D_MODEL + f] = f2bf(oacc[dt][r] / lrow[r]);
        }
    }
}

// ---------------------------------------------------------------------------
extern "C" void kernel_launch(void* const* d_in, const int* in_sizes, int n_in,
                              void* d_out, int out_size, void* d_ws, size_t ws_size,
                              hipStream_t stream) {
    (void)in_sizes; (void)n_in; (void)out_size; (void)ws_size;
    const float* x       = (const float*)d_in[0];
    const float* norm_w  = (const float*)d_in[1];
    const float* wq      = (const float*)d_in[2];
    const float* kv_down = (const float*)d_in[3];
    const float* wk      = (const float*)d_in[4];
    const float* wv      = (const float*)d_in[5];
    const float* wo      = (const float*)d_in[6];
    float* out = (float*)d_out;

    char* base = (char*)d_ws;
    size_t off = 0;
    auto alloc = [&](size_t bytes) {
        char* p = base + off;
        off = (off + bytes + 255) & ~(size_t)255;
        return p;
    };
    bf16raw* xn    = (bf16raw*)alloc((size_t)NTOK * D_MODEL * 2);
    bf16raw* wq_t  = (bf16raw*)alloc((size_t)D_MODEL * (NHEAD * HDIM) * 2);
    bf16raw* kvd_t = (bf16raw*)alloc((size_t)LAT * D_MODEL * 2);
    bf16raw* wk_t  = (bf16raw*)alloc((size_t)(NHEAD * HDIM) * LAT * 2);
    bf16raw* wv_t  = (bf16raw*)alloc((size_t)(NHEAD * HDIM) * LAT * 2);
    bf16raw* wo_t  = (bf16raw*)alloc((size_t)D_MODEL * D_MODEL * 2);
    bf16raw* qb    = (bf16raw*)alloc((size_t)NTOK * D_MODEL * 2);
    bf16raw* kvb   = (bf16raw*)alloc((size_t)NTOK * LAT * 2);
    bf16raw* kb_   = (bf16raw*)alloc((size_t)NTOK * D_MODEL * 2);
    bf16raw* vtb   = (bf16raw*)alloc((size_t)NTOK * D_MODEL * 2);
    bf16raw* aout  = (bf16raw*)alloc((size_t)NTOK * D_MODEL * 2);
    float*   cosT  = (float*)alloc((size_t)S_LEN * 32 * 4);
    float*   sinT  = (float*)alloc((size_t)S_LEN * 32 * 4);

    rope_table_kernel<<<(S_LEN * 32 + 255) / 256, 256, 0, stream>>>(cosT, sinT);
    rmsnorm_kernel<<<NTOK, 256, 0, stream>>>(x, norm_w, xn);

    cast_transpose_kernel<<<(D_MODEL * D_MODEL + 255) / 256, 256, 0, stream>>>(wq, wq_t, D_MODEL, D_MODEL);
    cast_transpose_kernel<<<(D_MODEL * LAT + 255) / 256, 256, 0, stream>>>(kv_down, kvd_t, D_MODEL, LAT);
    cast_transpose_kernel<<<(LAT * D_MODEL + 255) / 256, 256, 0, stream>>>(wk, wk_t, LAT, D_MODEL);
    cast_transpose_kernel<<<(LAT * D_MODEL + 255) / 256, 256, 0, stream>>>(wv, wv_t, LAT, D_MODEL);
    cast_transpose_kernel<<<(D_MODEL * D_MODEL + 255) / 256, 256, 0, stream>>>(wo, wo_t, D_MODEL, D_MODEL);

    dim3 blk(256);
    // q = RoPE(xn @ wq)
    gemm_kernel<1><<<dim3(D_MODEL / 128, NTOK / 128), blk, 0, stream>>>(
        xn, wq_t, NTOK, D_MODEL, D_MODEL, qb, nullptr, nullptr, cosT, sinT);
    // kv = xn @ kv_down
    gemm_kernel<0><<<dim3(LAT / 128, NTOK / 128), blk, 0, stream>>>(
        xn, kvd_t, NTOK, LAT, D_MODEL, kvb, nullptr, nullptr, nullptr, nullptr);
    // k = RoPE(kv @ wk)
    gemm_kernel<1><<<dim3(D_MODEL / 128, NTOK / 128), blk, 0, stream>>>(
        kvb, wk_t, NTOK, D_MODEL, LAT, kb_, nullptr, nullptr, cosT, sinT);
    // v^T = (kv @ wv)^T  -> (B,H,HD,S)
    gemm_kernel<3><<<dim3(D_MODEL / 128, NTOK / 128), blk, 0, stream>>>(
        kvb, wv_t, NTOK, D_MODEL, LAT, vtb, nullptr, nullptr, nullptr, nullptr);
    // attention
    attn_kernel<<<dim3(S_LEN / 128, BATCH * NHEAD), blk, 0, stream>>>(qb, kb_, vtb, aout);
    // out = x + attn @ wo
    gemm_kernel<2><<<dim3(D_MODEL / 128, NTOK / 128), blk, 0, stream>>>(
        aout, wo_t, NTOK, D_MODEL, D_MODEL, nullptr, out, x, nullptr, nullptr);
}
